// fullRNN_61632780698081
// MI455X (gfx1250) — matrix operand order repaired and checked by reading, not compiled, not run on hardware
//
#include <hip/hip_runtime.h>
#include <hip/hip_bf16.h>

typedef _Float16 f16;
typedef __attribute__((ext_vector_type(16))) _Float16 v16h;
typedef __attribute__((ext_vector_type(8)))  _Float16 v8h;
typedef __attribute__((ext_vector_type(8)))  float    v8f;

#define BSZ  128
#define TT   512
#define NN   512
#define CIN  8
#define COUT 8

static constexpr float DECAY  = 0.9f;                    // 1 - dt/tau
static constexpr float GAIN   = 0.1f;                    // dt/tau
static constexpr float NS_IN  = 0.014142135623730951f;   // sqrt(2)*g_in
static constexpr float NS_REC = 0.21213203435596426f;    // sqrt(2)*g_rec

// ---- prologue: pack W_w (fp32 [N][N]) -> f16 row-major [512][512] ----
__global__ void pack_w_f16(const float* __restrict__ W, f16* __restrict__ Wp, int n) {
    int i = blockIdx.x * blockDim.x + threadIdx.x;
    if (i < n) Wp[i] = (f16)W[i];
}

// ---- prologue: pack Out_w (fp32 [8][512]) -> f16 [16][512], rows 8..15 zero ----
__global__ void pack_outw_f16(const float* __restrict__ Ow, f16* __restrict__ Op) {
    int i = blockIdx.x * blockDim.x + threadIdx.x;
    if (i < 16 * NN) {
        int r = i >> 9;
        Op[i] = (r < COUT) ? (f16)Ow[i] : (f16)0.0f;
    }
}

// ---- persistent fused RNN scan: 8 blocks x 1024 threads (32 waves) ----
// wave w owns N-tile w (16 columns); 8 waves/SIMD for latency hiding.
__global__ __launch_bounds__(1024) void rnn_scan(
    const float* __restrict__ x,        const float* __restrict__ noise_in,
    const float* __restrict__ noise_rec,
    const float* __restrict__ In_w,     const float* __restrict__ In_b,
    const float* __restrict__ Out_b,    const float* __restrict__ h0,
    const f16*   __restrict__ Wp,       const f16*   __restrict__ Op,
    float* __restrict__ hidden,         float* __restrict__ outp)
{
    __shared__ __align__(16) f16 th[16 * NN];   // tanh(h) tile, f16, [m][n]
    __shared__ float Inw[NN * CIN];
    __shared__ float Inb[NN];
    __shared__ float xs[16 * CIN];              // noisy input frame for step t

    const int tid  = threadIdx.x;
    const int wave = tid >> 5;
    const int lane = tid & 31;
    const int hi   = lane >> 4;      // 0 = lanes 0-15, 1 = lanes 16-31
    const int ln   = lane & 15;
    const int b0   = blockIdx.x * 16;
    const int nt   = wave;           // one 16-wide N-tile per wave
    const int n    = nt * 16 + ln;   // this lane's output column (C/D layout)

    // one-time LDS init (covered by first barrier)
    if (tid < NN) {
        int nn = tid;
        Inb[nn] = In_b[nn];
        #pragma unroll
        for (int c = 0; c < CIN; ++c) Inw[nn * CIN + c] = In_w[nn * CIN + c];
        f16 t0h = (f16)tanhf(h0[nn]);
        #pragma unroll 4
        for (int m = 0; m < 16; ++m) th[m * NN + nn] = t0h;
    }
    if (tid < 16 * CIN) {   // xs for t = 0
        int m = tid >> 3, c = tid & 7;
        int gi = ((b0 + m) * TT + 0) * CIN + c;
        xs[tid] = x[gi] + NS_IN * noise_in[gi];
    }

    // persistent hidden state in VGPRs: element (m = r + 8*hi, n)
    float hreg[8];
    {
        float hv = h0[n];
        #pragma unroll
        for (int r = 0; r < 8; ++r) hreg[r] = hv;
    }
    float ob = 0.0f;
    if (wave == 0 && ln < COUT) ob = Out_b[ln];

    const f16* wrow = Wp + n * NN;       // B fragment row (W_w row n)
    const f16* orow = Op + ln * NN;      // Out_w row c (wave 0 only)
    const f16* thr  = th + ln * NN;      // A fragment row m = ln

    for (int t = 0; t < TT; ++t) {
        __syncthreads();   // th = tanh(h_{t-1}) and xs_t are ready

        // prefetch this step's recurrent noise (streams once; keep out of L2)
        float nrv[8];
        #pragma unroll
        for (int r = 0; r < 8; ++r) {
            const int m = r + 8 * hi;
            nrv[r] = __builtin_nontemporal_load(
                &noise_rec[((b0 + m) * TT + t) * NN + n]);
        }

        // ---- recurrent matmul: acc = tanh(h_{t-1}) @ W^T (this wave's tile) ----
        v8f acc = (v8f)0.0f;
        #pragma unroll 4
        for (int ks = 0; ks < 16; ++ks) {
            v16h a;
            ((v8h*)&a)[0] = *(const v8h*)(thr + ks * 32 + hi * 8);
            ((v8h*)&a)[1] = *(const v8h*)(thr + ks * 32 + 16 + hi * 8);
            v16h bv;
            ((v8h*)&bv)[0] = *(const v8h*)(wrow + ks * 32 + hi * 8);
            ((v8h*)&bv)[1] = *(const v8h*)(wrow + ks * 32 + 16 + hi * 8);
            acc = __builtin_amdgcn_wmma_f32_16x16x32_f16(
                false, a, false, bv, (short)0, acc, false, false);
        }

        // ---- output projection for step t-1 (wave 0): tanh(h_{t-1}) @ Out_w^T ----
        if (wave == 0 && t > 0) {
            v8f oacc = (v8f)0.0f;
            #pragma unroll 4
            for (int ks = 0; ks < 16; ++ks) {
                v16h a;
                ((v8h*)&a)[0] = *(const v8h*)(thr + ks * 32 + hi * 8);
                ((v8h*)&a)[1] = *(const v8h*)(thr + ks * 32 + 16 + hi * 8);
                v16h bo;
                ((v8h*)&bo)[0] = *(const v8h*)(orow + ks * 32 + hi * 8);
                ((v8h*)&bo)[1] = *(const v8h*)(orow + ks * 32 + 16 + hi * 8);
                oacc = __builtin_amdgcn_wmma_f32_16x16x32_f16(
                    false, a, false, bo, (short)0, oacc, false, false);
            }
            if (ln < COUT) {
                #pragma unroll
                for (int r = 0; r < 8; ++r) {
                    int m = r + 8 * hi;
                    outp[((b0 + m) * TT + (t - 1)) * COUT + ln] = oacc[r] + ob;
                }
            }
        }

        // ---- state update: h = decay*h + gain*(acc + u + ns*nr) ----
        const float ib = Inb[n];
        #pragma unroll
        for (int r = 0; r < 8; ++r) {
            const int m = r + 8 * hi;
            float u = ib;
            #pragma unroll
            for (int c = 0; c < CIN; ++c) u += xs[m * CIN + c] * Inw[n * CIN + c];
            float h = DECAY * hreg[r] + GAIN * (acc[r] + u + NS_REC * nrv[r]);
            hreg[r] = h;
            __builtin_nontemporal_store(h, &hidden[((b0 + m) * TT + t) * NN + n]);
        }

        __syncthreads();   // all reads of old th done

        // ---- publish tanh(h_t) and stage xs_{t+1} ----
        #pragma unroll
        for (int r = 0; r < 8; ++r) {
            const int m = r + 8 * hi;
            th[m * NN + n] = (f16)tanhf(hreg[r]);
        }
        if (tid < 16 * CIN && t + 1 < TT) {
            int m = tid >> 3, c = tid & 7;
            int gi = ((b0 + m) * TT + (t + 1)) * CIN + c;
            xs[tid] = x[gi] + NS_IN * noise_in[gi];
        }
    }

    // ---- epilogue: output projection for t = T-1 ----
    __syncthreads();
    if (wave == 0) {
        v8f oacc = (v8f)0.0f;
        #pragma unroll 4
        for (int ks = 0; ks < 16; ++ks) {
            v16h a;
            ((v8h*)&a)[0] = *(const v8h*)(thr + ks * 32 + hi * 8);
            ((v8h*)&a)[1] = *(const v8h*)(thr + ks * 32 + 16 + hi * 8);
            v16h bo;
            ((v8h*)&bo)[0] = *(const v8h*)(orow + ks * 32 + hi * 8);
            ((v8h*)&bo)[1] = *(const v8h*)(orow + ks * 32 + 16 + hi * 8);
            oacc = __builtin_amdgcn_wmma_f32_16x16x32_f16(
                false, a, false, bo, (short)0, oacc, false, false);
        }
        if (ln < COUT) {
            #pragma unroll
            for (int r = 0; r < 8; ++r) {
                int m = r + 8 * hi;
                outp[((b0 + m) * TT + (TT - 1)) * COUT + ln] = oacc[r] + ob;
            }
        }
    }
}

extern "C" void kernel_launch(void* const* d_in, const int* in_sizes, int n_in,
                              void* d_out, int out_size, void* d_ws, size_t ws_size,
                              hipStream_t stream) {
    (void)in_sizes; (void)n_in; (void)out_size; (void)ws_size;
    const float* x        = (const float*)d_in[0];
    const float* noise_in = (const float*)d_in[1];
    const float* noise_rec= (const float*)d_in[2];
    const float* In_w     = (const float*)d_in[3];
    const float* In_b     = (const float*)d_in[4];
    const float* W_w      = (const float*)d_in[5];
    const float* Out_w    = (const float*)d_in[6];
    const float* Out_b    = (const float*)d_in[7];
    const float* h0       = (const float*)d_in[8];

    f16* Wp = (f16*)d_ws;                 // 512*512 f16 = 512 KB
    f16* Op = Wp + (size_t)NN * NN;       // 16*512 f16  = 16 KB

    pack_w_f16<<<(NN * NN + 255) / 256, 256, 0, stream>>>(W_w, Wp, NN * NN);
    pack_outw_f16<<<(16 * NN + 255) / 256, 256, 0, stream>>>(Out_w, Op);

    float* hidden = (float*)d_out;                       // [B,T,N]
    float* outp   = hidden + (size_t)BSZ * TT * NN;      // [B,T,COUT]

    rnn_scan<<<BSZ / 16, 1024, 0, stream>>>(x, noise_in, noise_rec,
                                            In_w, In_b, Out_b, h0,
                                            Wp, Op, hidden, outp);
}
